// ODEFunc_45114336477555
// MI455X (gfx1250) — matrix operand order repaired and checked by reading, not compiled, not run on hardware
//
#include <hip/hip_runtime.h>

typedef __attribute__((ext_vector_type(16))) _Float16 v16h;
typedef __attribute__((ext_vector_type(8)))  _Float16 v8h;
typedef __attribute__((ext_vector_type(8)))  float    v8f;
typedef __attribute__((ext_vector_type(4)))  int      v4i;

#define B_      2
#define NN      50000
#define FDIM    13
#define EE      250000
#define MM      128
#define HH      128
#define EDGE_IN 34
#define EK      64      // padded K for edge layer-1 (34 -> 64)
#define NODE_IN 269
#define NK      288     // padded K for node layer-1 (269 -> 288)
#define EPB     128     // edges per block (4 chunks of 32)

#if __has_builtin(__builtin_amdgcn_global_load_async_to_lds_b128)
#define HAVE_ASYNC_LDS 1
#endif

// ---------------------------------------------------------------------------
// Fast transcendentals
// ---------------------------------------------------------------------------
__device__ inline float tanh_fast(float x) {
#if __has_builtin(__builtin_amdgcn_tanhf)
  return __builtin_amdgcn_tanhf(x);
#else
  return tanhf(x);
#endif
}

__device__ inline float sigmoid_fast(float x) {
#if __has_builtin(__builtin_amdgcn_rcpf)
  return __builtin_amdgcn_rcpf(1.0f + __expf(-x));
#else
  return 1.0f / (1.0f + __expf(-x));
#endif
}

// ---------------------------------------------------------------------------
// Async global -> LDS staging (CDNA5 ASYNCcnt path), with scalar fallback
// ---------------------------------------------------------------------------
__device__ inline void stage16(const _Float16* g, _Float16* l) {
#ifdef HAVE_ASYNC_LDS
  v4i* gv = const_cast<v4i*>((const v4i*)g);
  v4i* lv = (v4i*)l;
  __builtin_amdgcn_global_load_async_to_lds_b128(
      (__attribute__((address_space(1))) v4i*)gv,
      (__attribute__((address_space(3))) v4i*)lv, 0, 0);
#else
  *(v8h*)l = *(const v8h*)g;
#endif
}

__device__ inline void stage_weights(const _Float16* __restrict__ g,
                                     _Float16* __restrict__ l,
                                     int halfs, int tid) {
  for (int i = tid * 8; i < halfs; i += 256 * 8) stage16(g + i, l + i);
}

__device__ inline void stage_wait() {
#ifdef HAVE_ASYNC_LDS
#if __has_builtin(__builtin_amdgcn_s_wait_asynccnt)
  __builtin_amdgcn_s_wait_asynccnt(0);
#else
  asm volatile("s_wait_asynccnt 0x0" ::: "memory");
#endif
#endif
}

// ---------------------------------------------------------------------------
// Fragment helpers (wave32 WMMA 16x16x32 f16 layouts per CDNA5 ISA 7.12.2)
// ---------------------------------------------------------------------------
__device__ inline v16h load_a_frag(const _Float16* base, int pitch_h,
                                   int rowbase, int kt, int lane) {
  const int m = lane & 15;
  const int h = lane >> 4;
  const _Float16* p = base + (size_t)(rowbase + m) * pitch_h + kt * 32 + 8 * h;
  union { v16h v; v8h half[2]; } a;
  a.half[0] = *(const v8h*)(p);        // K = kt*32 + 8h .. +7
  a.half[1] = *(const v8h*)(p + 16);   // K = kt*32 + 16 + 8h .. +7
  return a.v;
}

__device__ inline v16h load_b_frag(const _Float16* wswz, int KT,
                                   int kt, int nt, int lane) {
  // pre-swizzled: one 32-byte contiguous chunk per lane
  const _Float16* p = wswz + ((size_t)(nt * KT + kt) << 9) + lane * 16;
  return *(const v16h*)p;
}

// ---------------------------------------------------------------------------
// Weight pre-swizzle: f32 row-major [K][N] -> f16 WMMA B-fragment layout
// ---------------------------------------------------------------------------
__global__ void swizzle_w(const float* __restrict__ src, _Float16* __restrict__ dst,
                          int K, int Nc, int KT, int NT) {
  int idx = blockIdx.x * blockDim.x + threadIdx.x;
  int total = KT * NT * 512;
  if (idx >= total) return;
  int blk = idx >> 9;
  int r   = idx & 511;
  int L   = r >> 4;
  int e   = r & 15;
  int kt  = blk % KT;
  int nt  = blk / KT;
  int k   = kt * 32 + ((e >> 3) << 4) + ((L >> 4) << 3) + (e & 7);
  int n   = nt * 16 + (L & 15);
  float v = (k < K && n < Nc) ? src[(size_t)k * Nc + n] : 0.0f;
  dst[idx] = (_Float16)v;
}

__global__ void zero_f32(float* __restrict__ p, size_t n) {
  size_t i = (size_t)blockIdx.x * blockDim.x + threadIdx.x;
  size_t s = (size_t)gridDim.x * blockDim.x;
  for (; i < n; i += s) p[i] = 0.0f;
}

// ---------------------------------------------------------------------------
// Edge kernel: 128 edges/block (4 chunks x 32), 8 waves.
// Weights staged to LDS once via async copy; WMMA layer1 (K=64) x2 branches,
// VALU sigmoid gate, WMMA layer2 (K=128), gated atomic scatter-add.
// ---------------------------------------------------------------------------
__global__ __launch_bounds__(256)
void edge_kernel(const float* __restrict__ z_all,   // (B,N,13)
                 const int*   __restrict__ edges,   // (B,2,E)
                 float*       __restrict__ agg,     // (B,N,128)
                 const _Float16* __restrict__ w1m, const float* __restrict__ b1m,
                 const _Float16* __restrict__ w2m, const float* __restrict__ b2m,
                 const _Float16* __restrict__ w1w, const float* __restrict__ b1w,
                 const float*    __restrict__ w2w, const float* __restrict__ b2w) {
  __shared__ __align__(32) _Float16 sW1m[2 * 8 * 512];   // 16 KB
  __shared__ __align__(32) _Float16 sW1w[2 * 8 * 512];   // 16 KB
  __shared__ __align__(32) _Float16 sW2m[4 * 8 * 512];   // 32 KB
  __shared__ __align__(32) _Float16 ef [32][EK];
  __shared__ __align__(32) _Float16 h1m[32][HH];
  __shared__ __align__(32) _Float16 h1w[32][HH];
  __shared__ float wsig[32];
  __shared__ int   tgt_l[32];
  __shared__ int   val_l[32];

  const int b   = blockIdx.y;
  const int tid = threadIdx.x;
  const float* z    = z_all + (size_t)b * NN * FDIM;
  const int*   srcs = edges + (size_t)b * 2 * EE;
  const int*   tgts = srcs + EE;
  float*       aggb = agg   + (size_t)b * NN * MM;

  // ---- stage all weight fragments into LDS (ASYNCcnt) ----
  stage_weights(w1m, sW1m, 2 * 8 * 512, tid);
  stage_weights(w1w, sW1w, 2 * 8 * 512, tid);
  stage_weights(w2m, sW2m, 4 * 8 * 512, tid);
  stage_wait();
  __syncthreads();

  const int wave = tid >> 5;
  const int lane = tid & 31;
  const int rt   = wave >> 2;   // row-tile (16 edges)
  const int cg   = wave & 3;    // column group
  const int rowbase = rt * 16;
  const int m_ = lane & 15;
  const int h_ = lane >> 4;

  for (int c = 0; c < EPB / 32; ++c) {
    const int e0 = blockIdx.x * EPB + c * 32;
    if (e0 >= EE) break;                       // block-uniform

    // ---- build edge features (one thread per edge) ----
    if (tid < 32) {
      int  e  = e0 + tid;
      bool ok = (e < EE);
      int  s  = ok ? srcs[e] : 0;
      int  t  = ok ? tgts[e] : 0;
      tgt_l[tid] = t;
      val_l[tid] = ok ? 1 : 0;
      float zs[FDIM], zt[FDIM];
#pragma unroll
      for (int i = 0; i < FDIM; ++i) {
        zs[i] = ok ? z[(size_t)s * FDIM + i] : 0.f;
        zt[i] = ok ? z[(size_t)t * FDIM + i] : 0.f;
      }
      float dx = zs[0]-zt[0], dy = zs[1]-zt[1], dz = zs[2]-zt[2];
      float dist = dx*dx + dy*dy + dz*dz;
      float ax = zs[3], ay = zs[4], az = zs[5];
      float bx = zt[3], by = zt[4], bz = zt[5];
      float cx = ay*bz - az*by, cy = az*bx - ax*bz, cz = ax*by - ay*bx;
      float cn = sqrtf(cx*cx + cy*cy + cz*cz);
      _Float16* row = ef[tid];
#pragma unroll
      for (int i = 0; i < FDIM; ++i) { row[i] = (_Float16)zs[i]; row[FDIM+i] = (_Float16)zt[i]; }
      row[26] = (_Float16)dx; row[27] = (_Float16)dy; row[28] = (_Float16)dz;
      row[29] = (_Float16)dist;
      row[30] = (_Float16)cx; row[31] = (_Float16)cy; row[32] = (_Float16)cz;
      row[33] = (_Float16)cn;
#pragma unroll
      for (int i = EDGE_IN; i < EK; ++i) row[i] = (_Float16)0.f;
    }
    __syncthreads();

    // ---- layer 1, message + gate branches, 2 N-tiles per wave ----
#pragma unroll
    for (int br = 0; br < 2; ++br) {
      const _Float16* w1 = br ? sW1w : sW1m;
      const float*    bb = br ? b1w : b1m;
      _Float16* dst = br ? &h1w[0][0] : &h1m[0][0];
#pragma unroll
      for (int j = 0; j < 2; ++j) {
        int nt = cg + j * 4;
        v8f acc = {};
#pragma unroll
        for (int kt = 0; kt < 2; ++kt) {
          v16h a  = load_a_frag(&ef[0][0], EK, rowbase, kt, lane);
          v16h bf = load_b_frag(w1, 2, kt, nt, lane);
          acc = __builtin_amdgcn_wmma_f32_16x16x32_f16(false, a, false, bf,
                                                       (short)0, acc, false, false);
        }
        int col = nt * 16 + m_;
        float bias = bb[col];
#pragma unroll
        for (int v = 0; v < 8; ++v)
          dst[(size_t)(rowbase + v + 8 * h_) * HH + col] =
              (_Float16)tanh_fast(acc[v] + bias);
      }
    }
    __syncthreads();

    // ---- sigmoid gate (128 -> 1), one thread per edge ----
    if (tid < 32) {
      float acc = 0.f;
#pragma unroll 8
      for (int k = 0; k < HH; ++k) acc += (float)h1w[tid][k] * w2w[k];
      wsig[tid] = sigmoid_fast(acc + b2w[0]);
    }
    __syncthreads();

    // ---- layer 2 message (K=128) + gated scatter-add ----
#pragma unroll
    for (int j = 0; j < 2; ++j) {
      int nt = cg + j * 4;
      v8f acc = {};
#pragma unroll
      for (int kt = 0; kt < 4; ++kt) {
        v16h a  = load_a_frag(&h1m[0][0], HH, rowbase, kt, lane);
        v16h bf = load_b_frag(sW2m, 4, kt, nt, lane);
        acc = __builtin_amdgcn_wmma_f32_16x16x32_f16(false, a, false, bf,
                                                     (short)0, acc, false, false);
      }
      int col = nt * 16 + m_;
      float bias = b2m[col];
#pragma unroll
      for (int v = 0; v < 8; ++v) {
        int rr = rowbase + v + 8 * h_;
        if (val_l[rr]) {
          float mv = acc[v] + bias;
          unsafeAtomicAdd(&aggb[(size_t)tgt_l[rr] * MM + col], wsig[rr] * mv);
        }
      }
    }
    __syncthreads();
  }
}

// ---------------------------------------------------------------------------
// Node kernel: 32 nodes/block; layer1 K=288 WMMA, layer2 128->13 WMMA.
// ---------------------------------------------------------------------------
__global__ __launch_bounds__(256)
void node_kernel(const float* __restrict__ z_all,
                 const float* __restrict__ agg_h,
                 const float* __restrict__ agg_w,
                 float*       __restrict__ out,    // (B,N,13)
                 const _Float16* __restrict__ w1s, const float* __restrict__ b1,
                 const _Float16* __restrict__ w2s, const float* __restrict__ b2) {
  __shared__ __align__(32) _Float16 xin[32][NK];
  __shared__ __align__(32) _Float16 hh [32][HH];

  const int b   = blockIdx.y;
  const int n0  = blockIdx.x * 32;
  const int tid = threadIdx.x;
  const float* z  = z_all + (size_t)b * NN * FDIM;
  const float* ah = agg_h + (size_t)b * NN * MM;
  const float* aw = agg_w + (size_t)b * NN * MM;

  for (int idx = tid; idx < 32 * NK; idx += 256) {
    int r = idx / NK;
    int k = idx - r * NK;
    int node = n0 + r;
    float v = 0.f;
    if (node < NN) {
      if      (k < FDIM)          v = z [(size_t)node * FDIM + k];
      else if (k < FDIM + MM)     v = ah[(size_t)node * MM + (k - FDIM)];
      else if (k < FDIM + 2*MM)   v = aw[(size_t)node * MM + (k - FDIM - MM)];
    }
    xin[r][k] = (_Float16)v;
  }
  __syncthreads();

  const int wave = tid >> 5;
  const int lane = tid & 31;
  const int rt   = wave >> 2;
  const int cg   = wave & 3;
  const int rowbase = rt * 16;
  const int m_ = lane & 15;
  const int h_ = lane >> 4;

#pragma unroll
  for (int j = 0; j < 2; ++j) {
    int nt = cg + j * 4;
    v8f acc = {};
#pragma unroll
    for (int kt = 0; kt < 9; ++kt) {
      v16h a  = load_a_frag(&xin[0][0], NK, rowbase, kt, lane);
      v16h bf = load_b_frag(w1s, 9, kt, nt, lane);
      acc = __builtin_amdgcn_wmma_f32_16x16x32_f16(false, a, false, bf,
                                                   (short)0, acc, false, false);
    }
    int col = nt * 16 + m_;
    float bias = b1[col];
#pragma unroll
    for (int v = 0; v < 8; ++v)
      hh[rowbase + v + 8 * h_][col] = (_Float16)tanh_fast(acc[v] + bias);
  }
  __syncthreads();

  if (cg == 0) {           // waves 0 and 4: one 16-wide output tile (13 used)
    v8f acc = {};
#pragma unroll
    for (int kt = 0; kt < 4; ++kt) {
      v16h a  = load_a_frag(&hh[0][0], HH, rowbase, kt, lane);
      v16h bf = load_b_frag(w2s, 4, kt, 0, lane);
      acc = __builtin_amdgcn_wmma_f32_16x16x32_f16(false, a, false, bf,
                                                   (short)0, acc, false, false);
    }
    if (m_ < FDIM) {
      float bias = b2[m_];
#pragma unroll
      for (int v = 0; v < 8; ++v) {
        int node = n0 + rowbase + v + 8 * h_;
        if (node < NN)
          out[((size_t)b * NN + node) * FDIM + m_] = acc[v] + bias;
      }
    }
  }
}

// ---------------------------------------------------------------------------
extern "C" void kernel_launch(void* const* d_in, const int* in_sizes, int n_in,
                              void* d_out, int out_size, void* d_ws, size_t ws_size,
                              hipStream_t stream) {
  const float* z_h    = (const float*)d_in[0];
  const int*   e_hh   = (const int*)  d_in[1];
  const int*   e_w    = (const int*)  d_in[2];
  const float* ew1 = (const float*)d_in[3],  *eb1 = (const float*)d_in[4];
  const float* ew2 = (const float*)d_in[5],  *eb2 = (const float*)d_in[6];
  const float* ww1 = (const float*)d_in[7],  *wb1 = (const float*)d_in[8];
  const float* ww2 = (const float*)d_in[9],  *wb2 = (const float*)d_in[10];
  const float* xw1 = (const float*)d_in[11], *xb1 = (const float*)d_in[12];
  const float* xw2 = (const float*)d_in[13], *xb2 = (const float*)d_in[14];
  const float* yw1 = (const float*)d_in[15], *yb1 = (const float*)d_in[16];
  const float* yw2 = (const float*)d_in[17], *yb2 = (const float*)d_in[18];
  const float* nw1 = (const float*)d_in[19], *nb1 = (const float*)d_in[20];
  const float* nw2 = (const float*)d_in[21], *nb2 = (const float*)d_in[22];
  float* out = (float*)d_out;

  // workspace layout
  const size_t AGG = (size_t)B_ * NN * MM;           // 12.8M floats each
  float* agg_h = (float*)d_ws;
  float* agg_w = agg_h + AGG;
  _Float16* wb = (_Float16*)(agg_w + AGG);
  _Float16* ew1s = wb;                    // KT=2 NT=8
  _Float16* ew2s = ew1s + 2*8*512;        // KT=4 NT=8
  _Float16* ww1s = ew2s + 4*8*512;        // KT=2 NT=8
  _Float16* xw1s = ww1s + 2*8*512;
  _Float16* xw2s = xw1s + 2*8*512;
  _Float16* yw1s = xw2s + 4*8*512;
  _Float16* nw1s = yw1s + 2*8*512;        // KT=9 NT=8
  _Float16* nw2s = nw1s + 9*8*512;        // KT=4 NT=1

  zero_f32<<<2048, 256, 0, stream>>>(agg_h, 2 * AGG);

  auto swz = [&](const float* s, _Float16* d, int K, int Nc, int KT, int NT) {
    int total = KT * NT * 512;
    swizzle_w<<<(total + 255) / 256, 256, 0, stream>>>(s, d, K, Nc, KT, NT);
  };
  swz(ew1, ew1s, EDGE_IN, HH, 2, 8);
  swz(ew2, ew2s, HH,      MM, 4, 8);
  swz(ww1, ww1s, EDGE_IN, HH, 2, 8);
  swz(xw1, xw1s, EDGE_IN, HH, 2, 8);
  swz(xw2, xw2s, HH,      MM, 4, 8);
  swz(yw1, yw1s, EDGE_IN, HH, 2, 8);
  swz(nw1, nw1s, NODE_IN, HH, 9, 8);
  swz(nw2, nw2s, HH,    FDIM, 4, 1);

  dim3 egrid((EE + EPB - 1) / EPB, B_);
  edge_kernel<<<egrid, 256, 0, stream>>>(z_h, e_hh, agg_h,
                                         ew1s, eb1, ew2s, eb2,
                                         ww1s, wb1, ww2, wb2);
  edge_kernel<<<egrid, 256, 0, stream>>>(z_h, e_w, agg_w,
                                         xw1s, xb1, xw2s, xb2,
                                         yw1s, yb1, yw2, yb2);

  dim3 ngrid((NN + 31) / 32, B_);
  node_kernel<<<ngrid, 256, 0, stream>>>(z_h, agg_h, agg_w, out,
                                         nw1s, nb1, nw2s, nb2);
}
